// Model_48653389529282
// MI455X (gfx1250) — matrix operand order repaired and checked by reading, not compiled, not run on hardware
//
#include <hip/hip_runtime.h>

#define DEVINL static __device__ __forceinline__

typedef __attribute__((ext_vector_type(16))) __bf16 bf16x16;
typedef __attribute__((ext_vector_type(8)))  float  f32x8;
typedef __attribute__((ext_vector_type(4)))  unsigned int u32x4;
typedef __attribute__((ext_vector_type(4)))  unsigned int tdm_g0_t;
typedef __attribute__((ext_vector_type(8)))  int          tdm_g1_t;
typedef __attribute__((ext_vector_type(4)))  int          tdm_g2_t;
typedef __attribute__((ext_vector_type(8)))  int          tdm_g4_t;

#define B_      8
#define SPB_    512
#define KNN_    64
#define M_      (B_*SPB_)
#define NGRID_  2025
#define RPAD_   2048
// 1/sqrt(1+1e-5) for the eval-mode BatchNorm
#define BNS_    0.9999950000374997f

DEVINL f32x8 wmma_bf16(bf16x16 a, bf16x16 b, f32x8 c) {
  return __builtin_amdgcn_wmma_f32_16x16x32_bf16(false, a, false, b, (short)0, c, false, false);
}

// A fragment: 16x32 bf16 tile, row-major in LDS, leading dim ld (elements, mult of 8).
// ISA 7.12.2: VGPR v<4 -> K = half*8 + {2v,2v+1}; v>=4 -> K = 16 + half*8 + {...}
// => per lane: 8 contiguous bf16 at k=half*8, 8 contiguous at k=16+half*8 -> 2x b128.
DEVINL bf16x16 load_a(const __bf16* base, int ld) {
  const int lane = threadIdx.x & 31;
  const __bf16* p = base + (lane & 15) * ld + ((lane >> 4) << 3);
  union { u32x4 q[2]; bf16x16 b; } u;
  u.q[0] = *(const u32x4*)p;
  u.q[1] = *(const u32x4*)(p + 16);
  return u.b;
}

// B fragment from pre-swizzled weights. A 32x16 (KxN) tile is stored as
// 16 rows (n) x 32 contiguous k-elements (1 KB per tile). Frag element i of
// lane (half,n) is k = half*8 + i (i<8), 16 + half*8 + (i-8) (i>=8) -> 2x global b128.
DEVINL bf16x16 load_b_sw(const __bf16* tile) {
  const int lane = threadIdx.x & 31;
  const __bf16* p = tile + ((lane & 15) << 5) + ((lane >> 4) << 3);
  union { u32x4 q[2]; bf16x16 b; } u;
  u.q[0] = *(const u32x4*)p;
  u.q[1] = *(const u32x4*)(p + 16);
  return u.b;
}

DEVINL float leaky(float x) { return x > 0.f ? x : 0.2f * x; }
DEVINL float relu_(float x) { return x > 0.f ? x : 0.f; }

// TDM: DMA a [rows x 32] bf16 tile (row stride ldg elems) from global into LDS,
// padding 12 DWORDs after every 16 DWORDs -> LDS row stride 56 elems (112 B).
DEVINL void tdm_tile_to_lds(unsigned lds_off, const __bf16* gptr, int rows, int ldg) {
  unsigned long long ga = (unsigned long long)(size_t)gptr;
  tdm_g0_t g0 = { 1u,                                   // count=1 descriptor
                  lds_off,                              // lds_addr
                  (unsigned)ga,
                  (unsigned)((ga >> 32) & 0x01ffffffu) | (2u << 30) }; // type=2
  // word0: data_size=1(2B) | pad_enable | pad_interval=3(16 dw) | pad_amount=11(12 dw)
  tdm_g1_t g1 = { (int)((1u << 16) | (1u << 20) | (3u << 22) | (11u << 25)),
                  (int)((unsigned)ldg << 16),           // tensor_dim0[15:0] @ bits 63:48
                  (int)(((unsigned)ldg >> 16) | ((unsigned)rows << 16)), // dim0 hi | dim1 lo
                  (int)(32u << 16),                     // dim1 hi | tile_dim0=32
                  rows,                                 // tile_dim1 | tile_dim2=0
                  ldg,                                  // tensor_dim0_stride lo
                  0, 0 };
  tdm_g2_t gz = { 0, 0, 0, 0 };
  tdm_g4_t g4 = { 0, 0, 0, 0, 0, 0, 0, 0 };
  __builtin_amdgcn_tensor_load_to_lds(g0, g1, gz, gz, g4, 0);
}

// ---------------------------------------------------------------- weight cvt+swizzle
// fp32 [K,N] row-major -> bf16 WMMA-B tile order: tile (kt,nt) of 32x16 stored
// as [n][k] (16 rows x 32 elems contiguous), tiles kt-major then nt.
__global__ void cvt_swizzle_kernel(const float* __restrict__ s, __bf16* __restrict__ d,
                                   int K, int N) {
  int idx = blockIdx.x * blockDim.x + threadIdx.x;
  if (idx >= K * N) return;
  int k = idx / N, n = idx - k * N;
  int kt = k >> 5, kk = k & 31, nt = n >> 4, nn = n & 15;
  d[((size_t)(kt * (N >> 4) + nt) << 9) + nn * 32 + kk] = (__bf16)s[idx];
}

// ---------------------------------------------------------------- PointConv
__global__ __launch_bounds__(256)
void pointconv_kernel(const float* __restrict__ pos,
                      const int* __restrict__ sample_idx,
                      const int* __restrict__ edge_x,
                      const float* __restrict__ cw0, const float* __restrict__ cb0,
                      const float* __restrict__ cg0, const float* __restrict__ cbeta0,
                      const __bf16* __restrict__ cw1sw,
                      const float* __restrict__ cb1, const float* __restrict__ cg1,
                      const float* __restrict__ cbeta1,
                      const __bf16* __restrict__ cw2sw,
                      const float* __restrict__ cb2, const float* __restrict__ cg2,
                      const float* __restrict__ cbeta2,
                      __bf16* __restrict__ featbf) {
  __shared__ float  sh_rel[64][4];
  __shared__ __align__(16) __bf16 sh_h1[64][72];
  __shared__ __align__(16) __bf16 sh_h2[64][136];
  const int mC = blockIdx.x;
  const int tid = threadIdx.x;
  const int wv = tid >> 5, lane = tid & 31;

  if (tid < 64) {
    int e  = mC * KNN_ + tid;
    int nb = edge_x[e];
    int sc = sample_idx[mC];
    sh_rel[tid][0] = pos[nb*3+0] - pos[sc*3+0];
    sh_rel[tid][1] = pos[nb*3+1] - pos[sc*3+1];
    sh_rel[tid][2] = pos[nb*3+2] - pos[sc*3+2];
  }
  __syncthreads();

  { // stage1: 3 -> 64, fp32
    int j = tid & 63, mb = tid >> 6;
    float w0 = cw0[j], w1 = cw0[64+j], w2 = cw0[128+j];
    float bb = cb0[j], scl = cg0[j]*BNS_, be = cbeta0[j];
#pragma unroll
    for (int i = 0; i < 16; ++i) {
      int m = i*4 + mb;
      float a = sh_rel[m][0]*w0 + sh_rel[m][1]*w1 + sh_rel[m][2]*w2 + bb;
      sh_h1[m][j] = (__bf16)leaky(a*scl + be);
    }
  }
  __syncthreads();

  { // stage2: [64,64] @ [64,128]; wave wv owns col tile wv (8 n-tiles)
    f32x8 acc[4] = {};
#pragma unroll
    for (int ks = 0; ks < 2; ++ks) {
      bf16x16 bf = load_b_sw(cw1sw + ((size_t)(ks*8 + wv) << 9));
#pragma unroll
      for (int rt = 0; rt < 4; ++rt) {
        bf16x16 af = load_a(&sh_h1[rt*16][ks*32], 72);
        acc[rt] = wmma_bf16(af, bf, acc[rt]);
      }
    }
    int n = wv*16 + (lane & 15);
    float scl = cg1[n]*BNS_, be = cbeta1[n], bb = cb1[n];
    int mh = (lane >> 4) * 8;
#pragma unroll
    for (int rt = 0; rt < 4; ++rt)
#pragma unroll
      for (int r = 0; r < 8; ++r)
        sh_h2[rt*16 + r + mh][n] = (__bf16)leaky((acc[rt][r] + bb)*scl + be);
  }
  __syncthreads();

  { // stage3: [64,128] @ [128,512] + column max over the 64 rows (32 n-tiles)
#pragma unroll
    for (int c4 = 0; c4 < 4; ++c4) {
      int ct = wv*4 + c4;
      f32x8 acc[4] = {};
#pragma unroll
      for (int ks = 0; ks < 4; ++ks) {
        bf16x16 bf = load_b_sw(cw2sw + ((size_t)((ks << 5) + ct) << 9));
#pragma unroll
        for (int rt = 0; rt < 4; ++rt) {
          bf16x16 af = load_a(&sh_h2[rt*16][ks*32], 136);
          acc[rt] = wmma_bf16(af, bf, acc[rt]);
        }
      }
      int n = ct*16 + (lane & 15);
      float scl = cg2[n]*BNS_, be = cbeta2[n], bb = cb2[n];
      float cmax = -3.0e38f;
#pragma unroll
      for (int rt = 0; rt < 4; ++rt)
#pragma unroll
        for (int r = 0; r < 8; ++r)
          cmax = fmaxf(cmax, leaky((acc[rt][r] + bb)*scl + be));
      cmax = fmaxf(cmax, __shfl_xor(cmax, 16, 32));
      if (lane < 16) featbf[mC*512 + n] = (__bf16)cmax;
    }
  }
}

// ---------------------------------------------------------------- Encoder L1
__global__ __launch_bounds__(256)
void enc1_kernel(const __bf16* __restrict__ featbf, const __bf16* __restrict__ ew0sw,
                 const float* __restrict__ ew0,
                 const float* __restrict__ eb0, const float* __restrict__ eg0,
                 const float* __restrict__ ebeta0,
                 const float* __restrict__ pos, const int* __restrict__ sample_idx,
                 __bf16* __restrict__ z1bf) {
  __shared__ __align__(16) __bf16 shA[64][56];
  const int tid = threadIdx.x;
  const int wv = tid >> 5, lane = tid & 31;
  const int m0 = blockIdx.y * 64, n0t = blockIdx.x * 4;
  const int rt = wv & 3;
  const unsigned lds_off = (unsigned)(size_t)&shA[0][0];
  f32x8 acc[2] = {};
  for (int kc = 0; kc < 16; ++kc) {
    if (wv == 0) {  // wave 0 DMAs the 64x32 A-chunk into padded LDS
      tdm_tile_to_lds(lds_off, featbf + (size_t)m0*512 + kc*32, 64, 512);
      __builtin_amdgcn_s_wait_tensorcnt(0);
    }
    __syncthreads();
    bf16x16 af = load_a(&shA[rt*16][0], 56);
#pragma unroll
    for (int j = 0; j < 2; ++j) {
      int ct = (wv >> 2)*2 + j;
      bf16x16 bf = load_b_sw(ew0sw + ((size_t)(kc*32 + n0t + ct) << 9));
      acc[j] = wmma_bf16(af, bf, acc[j]);
    }
    __syncthreads();
  }
  int mh = (lane >> 4)*8;
#pragma unroll
  for (int j = 0; j < 2; ++j) {
    int ct = (wv >> 2)*2 + j;
    int n = (n0t + ct)*16 + (lane & 15);
    float bb = eb0[n], scl = eg0[n]*BNS_, be = ebeta0[n];
    float wp0 = ew0[512*512 + n], wp1 = ew0[513*512 + n], wp2 = ew0[514*512 + n];
#pragma unroll
    for (int r = 0; r < 8; ++r) {
      int m = m0 + rt*16 + r + mh;
      int si = sample_idx[m];
      float v = acc[j][r] + bb + pos[si*3+0]*wp0 + pos[si*3+1]*wp1 + pos[si*3+2]*wp2;
      z1bf[m*512 + n] = (__bf16)leaky(v*scl + be);
    }
  }
}

// ---------------------------------------------------------------- Encoder L2
__global__ __launch_bounds__(256)
void enc2_kernel(const __bf16* __restrict__ z1bf, const __bf16* __restrict__ ew1sw,
                 const float* __restrict__ eb1, float* __restrict__ z) {
  __shared__ __align__(16) __bf16 shA[64][56];
  const int tid = threadIdx.x;
  const int wv = tid >> 5, lane = tid & 31;
  const int m0 = blockIdx.y * 64, n0t = blockIdx.x * 4;
  const int rt = wv & 3;
  const unsigned lds_off = (unsigned)(size_t)&shA[0][0];
  f32x8 acc[2] = {};
  for (int kc = 0; kc < 16; ++kc) {
    if (wv == 0) {
      tdm_tile_to_lds(lds_off, z1bf + (size_t)m0*512 + kc*32, 64, 512);
      __builtin_amdgcn_s_wait_tensorcnt(0);
    }
    __syncthreads();
    bf16x16 af = load_a(&shA[rt*16][0], 56);
#pragma unroll
    for (int j = 0; j < 2; ++j) {
      int ct = (wv >> 2)*2 + j;
      bf16x16 bf = load_b_sw(ew1sw + ((size_t)(kc*128 + n0t + ct) << 9));  // N=2048 -> 128 n-tiles
      acc[j] = wmma_bf16(af, bf, acc[j]);
    }
    __syncthreads();
  }
  int mh = (lane >> 4)*8;
#pragma unroll
  for (int j = 0; j < 2; ++j) {
    int ct = (wv >> 2)*2 + j;
    int n = (n0t + ct)*16 + (lane & 15);
    float bb = eb1[n];
#pragma unroll
    for (int r = 0; r < 8; ++r) {
      int m = m0 + rt*16 + r + mh;
      z[(size_t)m*2048 + n] = acc[j][r] + bb;
    }
  }
}

// ---------------------------------------------------------------- score / latent
__global__ void score_kernel(const float* __restrict__ z, float* __restrict__ score) {
  int m = blockIdx.x, tid = threadIdx.x;
  __shared__ float red[256];
  float s = 0.f;
  for (int c = tid; c < 1024; c += 256) s += expf(-0.5f * z[(size_t)m*2048 + 1024 + c]);
  red[tid] = s; __syncthreads();
  for (int w = 128; w > 0; w >>= 1) { if (tid < w) red[tid] += red[tid+w]; __syncthreads(); }
  if (tid == 0) score[m] = red[0] * (1.0f/1024.0f);
}

__global__ void latent_kernel(const float* __restrict__ z, const float* __restrict__ score,
                              float* __restrict__ opt) {
  int b = blockIdx.x, tid = threadIdx.x;
  __shared__ float ssc[512];
  __shared__ float rv[256];
  __shared__ int   ri[256];
  __shared__ int   topi[16];
  for (int i = tid; i < 512; i += 256) ssc[i] = score[b*512 + i];
  __syncthreads();
  for (int t = 0; t < 16; ++t) {   // iterative argmax, lowest-index tie-break
    float bv = -3.0e38f; int bi = 0x7fffffff;
    for (int i = tid; i < 512; i += 256) {
      float v = ssc[i];
      if (v > bv || (v == bv && i < bi)) { bv = v; bi = i; }
    }
    rv[tid] = bv; ri[tid] = bi; __syncthreads();
    for (int w = 128; w > 0; w >>= 1) {
      if (tid < w) {
        if (rv[tid+w] > rv[tid] || (rv[tid+w] == rv[tid] && ri[tid+w] < ri[tid])) {
          rv[tid] = rv[tid+w]; ri[tid] = ri[tid+w];
        }
      }
      __syncthreads();
    }
    if (tid == 0) { topi[t] = ri[0]; ssc[ri[0]] = -3.0e38f; }
    __syncthreads();
  }
  for (int c = tid; c < 1024; c += 256) {
    float num = 0.f, den = 0.f;
#pragma unroll
    for (int t = 0; t < 16; ++t) {
      size_t m = (size_t)(b*512 + topi[t]);
      float inv = expf(-0.5f * z[m*2048 + 1024 + c]);
      num += z[m*2048 + c] * inv; den += inv;
    }
    opt[b*1024 + c] = num / den;
  }
}

// ---------------------------------------------------------------- decoder base
__global__ __launch_bounds__(256)
void fold_base_kernel(const float* __restrict__ opt, const float* __restrict__ W,
                      const float* __restrict__ bias, float* __restrict__ base) {
  int b = blockIdx.y, n = blockIdx.x*256 + threadIdx.x;
  __shared__ float shc[1024];
  for (int i = threadIdx.x; i < 1024; i += 256) shc[i] = opt[b*1024 + i];
  __syncthreads();
  float acc = bias[n];
  for (int k = 0; k < 1024; ++k) acc += shc[k] * W[k*512 + n];
  base[b*512 + n] = acc;
}

// ---------------------------------------------------------------- decoder mid
__global__ __launch_bounds__(256)
void fold_mid_kernel(int mode, const float* __restrict__ base,
                     const float* __restrict__ Wex,          // W0 rows 1024.. (ld 512)
                     const float* __restrict__ o1buf,        // [B*RPAD,3] (mode 1)
                     const __bf16* __restrict__ W2sw, const float* __restrict__ b2,
                     __bf16* __restrict__ h2out) {
  __shared__ float  sh_ex[64][3];
  __shared__ __align__(16) __bf16 shA[64][56];
  const int tid = threadIdx.x;
  const int wv = tid >> 5, lane = tid & 31;
  const int b = blockIdx.y, rb = blockIdx.x;
  const int nex = (mode == 0) ? 2 : 3;
  if (tid < 64) {
    int g = rb*64 + tid;
    if (mode == 0) {
      int gp = g > 2024 ? 2024 : g;           // padded rows: clamp (unused downstream)
      sh_ex[tid][0] = -0.3f + 0.6f * (float)(gp % 45) * (1.0f/44.0f);
      sh_ex[tid][1] = -0.3f + 0.6f * (float)(gp / 45) * (1.0f/44.0f);
      sh_ex[tid][2] = 0.0f;
    } else {
      int rp = b*RPAD_ + g;
      sh_ex[tid][0] = o1buf[rp*3+0];
      sh_ex[tid][1] = o1buf[rp*3+1];
      sh_ex[tid][2] = o1buf[rp*3+2];
    }
  }
  __syncthreads();

  f32x8 acc[4][4] = {};
  const float* basep = base + b*512;
  for (int kc = 0; kc < 16; ++kc) {
    {
      int r = tid >> 2, kk0 = (tid & 3) * 8;
#pragma unroll
      for (int i = 0; i < 8; ++i) {
        int kk = kk0 + i, k = kc*32 + kk;
        float v = basep[k];
        for (int d = 0; d < nex; ++d) v += sh_ex[r][d] * Wex[d*512 + k];
        shA[r][kk] = (__bf16)relu_(v);
      }
    }
    __syncthreads();
    bf16x16 af[4];
#pragma unroll
    for (int rt = 0; rt < 4; ++rt) af[rt] = load_a(&shA[rt*16][0], 56);
#pragma unroll
    for (int c4 = 0; c4 < 4; ++c4) {
      bf16x16 bfr = load_b_sw(W2sw + ((size_t)(kc*32 + wv*4 + c4) << 9));
#pragma unroll
      for (int rt = 0; rt < 4; ++rt)
        acc[rt][c4] = wmma_bf16(af[rt], bfr, acc[rt][c4]);
    }
    __syncthreads();
  }
  int mh = (lane >> 4)*8;
#pragma unroll
  for (int c4 = 0; c4 < 4; ++c4) {
    int n = wv*64 + c4*16 + (lane & 15);
    float bb = b2[n];
#pragma unroll
    for (int rt = 0; rt < 4; ++rt)
#pragma unroll
      for (int r = 0; r < 8; ++r) {
        int rp = b*RPAD_ + rb*64 + rt*16 + r + mh;
        h2out[(size_t)rp*512 + n] = (__bf16)relu_(acc[rt][c4][r] + bb);
      }
  }
}

// ---------------------------------------------------------------- decoder head (512->3)
__global__ void fold_out_kernel(int mode, const __bf16* __restrict__ h2,
                                const float* __restrict__ W3, const float* __restrict__ b3,
                                float* __restrict__ outp) {
  int idx = blockIdx.x*blockDim.x + threadIdx.x;
  if (idx >= B_*RPAD_*3) return;
  int r = idx / 3, d = idx - r*3;
  float acc = 0.f;
  for (int n = 0; n < 512; ++n) acc += (float)h2[(size_t)r*512 + n] * W3[n*3 + d];
  acc += b3[d];
  if (mode == 0) outp[r*3 + d] = acc;
  else {
    int b = r >> 11, g = r & (RPAD_-1);
    if (g < NGRID_) outp[(b*NGRID_ + g)*3 + d] = acc;
  }
}

// ---------------------------------------------------------------- launcher
extern "C" void kernel_launch(void* const* d_in, const int* in_sizes, int n_in,
                              void* d_out, int out_size, void* d_ws, size_t ws_size,
                              hipStream_t stream) {
  (void)in_sizes; (void)n_in; (void)out_size; (void)ws_size;
  const float* pos    = (const float*)d_in[0];
  const float* cw0    = (const float*)d_in[1];
  const float* cb0    = (const float*)d_in[2];
  const float* cg0    = (const float*)d_in[3];
  const float* cbeta0 = (const float*)d_in[4];
  const float* cw1    = (const float*)d_in[5];
  const float* cb1    = (const float*)d_in[6];
  const float* cg1    = (const float*)d_in[7];
  const float* cbeta1 = (const float*)d_in[8];
  const float* cw2    = (const float*)d_in[9];
  const float* cb2    = (const float*)d_in[10];
  const float* cg2    = (const float*)d_in[11];
  const float* cbeta2 = (const float*)d_in[12];
  const float* ew0    = (const float*)d_in[13];
  const float* eb0    = (const float*)d_in[14];
  const float* eg0    = (const float*)d_in[15];
  const float* ebeta0 = (const float*)d_in[16];
  const float* ew1    = (const float*)d_in[17];
  const float* eb1    = (const float*)d_in[18];
  const float* f1w0   = (const float*)d_in[19];
  const float* f1b0   = (const float*)d_in[20];
  const float* f1w1   = (const float*)d_in[21];
  const float* f1b1   = (const float*)d_in[22];
  const float* f1w2   = (const float*)d_in[23];
  const float* f1b2   = (const float*)d_in[24];
  const float* f2w0   = (const float*)d_in[25];
  const float* f2b0   = (const float*)d_in[26];
  const float* f2w1   = (const float*)d_in[27];
  const float* f2b1   = (const float*)d_in[28];
  const float* f2w2   = (const float*)d_in[29];
  const float* f2b2   = (const float*)d_in[30];
  const int* sample_idx = (const int*)d_in[32];
  const int* edge_x     = (const int*)d_in[33];
  float* out = (float*)d_out;

  char* wsp = (char*)d_ws;
  size_t off = 0;
  auto alloc = [&](size_t bytes) -> void* {
    void* p = (void*)(wsp + off);
    off = (off + bytes + 255) & ~(size_t)255;
    return p;
  };
  __bf16* cw1sw  = (__bf16*)alloc((size_t)64*128*2);
  __bf16* cw2sw  = (__bf16*)alloc((size_t)128*512*2);
  __bf16* ew0sw  = (__bf16*)alloc((size_t)512*512*2);
  __bf16* ew1sw  = (__bf16*)alloc((size_t)512*2048*2);
  __bf16* f1w1sw = (__bf16*)alloc((size_t)512*512*2);
  __bf16* f2w1sw = (__bf16*)alloc((size_t)512*512*2);
  __bf16* featbf = (__bf16*)alloc((size_t)M_*512*2);
  __bf16* z1bf   = (__bf16*)alloc((size_t)M_*512*2);
  float*  z      = (float*) alloc((size_t)M_*2048*4);
  float*  score  = (float*) alloc((size_t)M_*4);
  float*  opt    = (float*) alloc((size_t)B_*1024*4);
  float*  base1  = (float*) alloc((size_t)B_*512*4);
  float*  base2  = (float*) alloc((size_t)B_*512*4);
  __bf16* h2buf  = (__bf16*)alloc((size_t)B_*RPAD_*512*2);
  float*  o1buf  = (float*) alloc((size_t)B_*RPAD_*3*4);

  auto cvtsw = [&](const float* s, __bf16* dptr, int K, int N) {
    cvt_swizzle_kernel<<<(K*N + 255)/256, 256, 0, stream>>>(s, dptr, K, N);
  };
  cvtsw(cw1,  cw1sw,  64,  128);
  cvtsw(cw2,  cw2sw,  128, 512);
  cvtsw(ew0,  ew0sw,  512, 512);       // first 512 rows of [515,512]
  cvtsw(ew1,  ew1sw,  512, 2048);
  cvtsw(f1w1, f1w1sw, 512, 512);
  cvtsw(f2w1, f2w1sw, 512, 512);

  pointconv_kernel<<<M_, 256, 0, stream>>>(pos, sample_idx, edge_x,
                                           cw0, cb0, cg0, cbeta0,
                                           cw1sw, cb1, cg1, cbeta1,
                                           cw2sw, cb2, cg2, cbeta2, featbf);
  enc1_kernel<<<dim3(8, 64), 256, 0, stream>>>(featbf, ew0sw, ew0, eb0, eg0, ebeta0,
                                               pos, sample_idx, z1bf);
  enc2_kernel<<<dim3(32, 64), 256, 0, stream>>>(z1bf, ew1sw, eb1, z);
  score_kernel<<<M_, 256, 0, stream>>>(z, score);
  latent_kernel<<<B_, 256, 0, stream>>>(z, score, opt);
  fold_base_kernel<<<dim3(2, B_), 256, 0, stream>>>(opt, f1w0, f1b0, base1);
  fold_base_kernel<<<dim3(2, B_), 256, 0, stream>>>(opt, f2w0, f2b0, base2);
  fold_mid_kernel<<<dim3(32, B_), 256, 0, stream>>>(0, base1, f1w0 + 1024*512, (const float*)nullptr,
                                                    f1w1sw, f1b1, h2buf);
  fold_out_kernel<<<(B_*RPAD_*3+255)/256, 256, 0, stream>>>(0, h2buf, f1w2, f1b2, o1buf);
  fold_mid_kernel<<<dim3(32, B_), 256, 0, stream>>>(1, base2, f2w0 + 1024*512, o1buf,
                                                    f2w1sw, f2b1, h2buf);
  fold_out_kernel<<<(B_*RPAD_*3+255)/256, 256, 0, stream>>>(1, h2buf, f2w2, f2b2, out);
}
